// CrossModalAttention_67568425501408
// MI455X (gfx1250) — matrix operand order repaired and checked by reading, not compiled, not run on hardware
//
#include <hip/hip_runtime.h>
#include <hip/hip_bf16.h>

typedef _Float16 half_t;
typedef __attribute__((ext_vector_type(16))) _Float16 v16h;
typedef __attribute__((ext_vector_type(8)))  _Float16 v8h;
typedef __attribute__((ext_vector_type(8)))  float    v8f;

#define B_    512
#define NTOK  343
#define Cdim  96
#define NH    3
#define HD    32
#define NP    384      // padded token stride for Q/K/Vt (6 query blocks of 64)
#define KT    352      // padded key count (22 tiles of 16)
#define NW    64
#define QSCALE 0.17677669529663687f   // 1/sqrt(32)

union V16u { v16h v; v8h h[2]; };

static __device__ inline v16h pack2(v8h lo, v8h hi) { V16u u; u.h[0] = lo; u.h[1] = hi; return u.v; }

static __device__ inline v8f wmma_f16(v16h a, v16h b, v8f c) {
  return __builtin_amdgcn_wmma_f32_16x16x32_f16(false, a, false, b, (short)0, c, false, false);
}

// ---------------------------------------------------------------------------
// Kernel W: one-shot f32 -> f16 conversion of qkv_w (288x96) and proj_w (96x96)
// so the GEMM kernels load B-operands directly as packed f16 (no inline cvt).
// ---------------------------------------------------------------------------
#define WQN (288 * Cdim)
#define WPN (Cdim * Cdim)
__global__ __launch_bounds__(256) void k_cvtw(const float* __restrict__ qkv_w,
                                              const float* __restrict__ proj_w,
                                              half_t* __restrict__ wh,
                                              half_t* __restrict__ ph) {
  int i = blockIdx.x * 256 + threadIdx.x;
  if (i < WQN) wh[i] = (half_t)qkv_w[i];
  else if (i < WQN + WPN) ph[i - WQN] = (half_t)proj_w[i - WQN];
}

// ---------------------------------------------------------------------------
// Kernel 0: comb[w][h][q][k(352)] = bias_table[rel_index[q,k], h] + mask[w,q,k]
//           pad keys (k >= 343) get -30000 so softmax kills them for free.
// ---------------------------------------------------------------------------
__global__ __launch_bounds__(KT) void k_comb(const float* __restrict__ mask,
                                             const float* __restrict__ table,
                                             const int* __restrict__ relidx,
                                             half_t* __restrict__ comb) {
  int idx = blockIdx.x;                 // w*NH*NTOK + h*NTOK + q
  int w   = idx / (NH * NTOK);
  int rem = idx - w * (NH * NTOK);
  int h   = rem / NTOK;
  int q   = rem - h * NTOK;
  int k   = threadIdx.x;                // 0..351
  float v;
  if (k < NTOK) {
    int ri = relidx[q * NTOK + k];
    v = table[ri * NH + h] + mask[((size_t)w * NTOK + q) * NTOK + k];
  } else {
    v = -30000.0f;
  }
  comb[((size_t)(w * NH + h) * NTOK + q) * KT + k] = (half_t)v;
}

// ---------------------------------------------------------------------------
// Kernel 1: fused QKV projection for both modalities.
//   out[m,f] = sum_k x[m,k]*qkv_w[f,k] + qkv_b[f]   (M per-batch padded to 352)
//   q-part scaled by 1/sqrt(hd).  Scatter:
//     Q,K : [mod][b][h][tok(NP)][32]  f16
//     Vt  : [mod][b][h][d][tok(NP)]   f16 (transposed, packed 16B stores)
// 288 threads = 9 waves, each wave owns 2 of the 18 feature tiles.
// ---------------------------------------------------------------------------
__global__ __launch_bounds__(288) void k_qkv(const float* __restrict__ x0,
                                             const float* __restrict__ x1,
                                             const half_t* __restrict__ wh,
                                             const float* __restrict__ bias,
                                             half_t* __restrict__ Q,
                                             half_t* __restrict__ K,
                                             half_t* __restrict__ Vt) {
  const int mtile = blockIdx.x;       // 0..21 (token tiles within one batch)
  const int b     = blockIdx.y;       // 0..511
  const int mod   = blockIdx.z;       // 0..1
  const float* x  = mod ? x1 : x0;
  const size_t TQ = (size_t)B_ * NH * NP * HD;
  half_t* Qm = Q  + (size_t)mod * TQ;
  half_t* Km = K  + (size_t)mod * TQ;
  half_t* Vm = Vt + (size_t)mod * TQ;

  __shared__ __align__(16) half_t As[16 * Cdim];
  const int tid = threadIdx.x;
  for (int i = tid; i < 16 * Cdim; i += 288) {
    int row = i / Cdim, col = i - row * Cdim;
    int tok = mtile * 16 + row;
    float v = (tok < NTOK) ? x[((size_t)b * NTOK + tok) * Cdim + col] : 0.0f;
    As[i] = (half_t)v;
  }
  __syncthreads();

  const int lane = tid & 31;
  const int wv   = tid >> 5;          // 0..8
  const int hf   = lane >> 4;         // K-chunk half
  const int ln   = lane & 15;

  v16h a[3];
#pragma unroll
  for (int kc = 0; kc < 3; kc++) {
    const half_t* pr = &As[ln * Cdim + kc * 32 + 16 * hf];
    a[kc] = pack2(*(const v8h*)pr, *(const v8h*)(pr + 8));
  }

  for (int t = 0; t < 2; t++) {
    const int ft = wv * 2 + t;              // 0..17
    const int f  = ft * 16 + ln;            // 0..287
    v8f acc = {};
#pragma unroll
    for (int kc = 0; kc < 3; kc++) {
      v16h bm = *(const v16h*)(wh + (size_t)f * Cdim + kc * 32 + 16 * hf);
      acc = wmma_f16(a[kc], bm, acc);
    }
    const float bv = bias[f];
    if (f < 96) {                           // q (scaled)
      int h = f >> 5, d = f & 31;
#pragma unroll
      for (int r = 0; r < 8; r++) {
        int tok = mtile * 16 + r + 8 * hf;
        if (tok < NTOK)
          Qm[(((size_t)b * NH + h) * NP + tok) * HD + d] = (half_t)((acc[r] + bv) * QSCALE);
      }
    } else if (f < 192) {                   // k
      int f2 = f - 96, h = f2 >> 5, d = f2 & 31;
#pragma unroll
      for (int r = 0; r < 8; r++) {
        int tok = mtile * 16 + r + 8 * hf;
        if (tok < NTOK)
          Km[(((size_t)b * NH + h) * NP + tok) * HD + d] = (half_t)(acc[r] + bv);
      }
    } else {                                // v -> transposed, packed 16B store
      int f2 = f - 192, h = f2 >> 5, d = f2 & 31;
      v8h pk;
#pragma unroll
      for (int r = 0; r < 8; r++) pk[r] = (half_t)(acc[r] + bv);
      int tok0 = mtile * 16 + 8 * hf;       // 8 consecutive tokens
      *(v8h*)&Vm[(((size_t)b * NH + h) * HD + d) * NP + tok0] = pk;
    }
  }
}

// ---------------------------------------------------------------------------
// Kernel 2: attention for one (query-block of 64, head, batch, branch).
//   S = QK^T (WMMA) -> LDS; softmax pass adds comb (coalesced f16 reads),
//   P f16 in LDS; O = P*V (WMMA, K=352 in 11 steps); 1/sum in epilogue.
// ---------------------------------------------------------------------------
__global__ __launch_bounds__(256) void k_attn(const half_t* __restrict__ Q,
                                              const half_t* __restrict__ K,
                                              const half_t* __restrict__ Vt,
                                              const half_t* __restrict__ comb,
                                              half_t* __restrict__ attn_out) {
  const int qb = blockIdx.x;              // 0..5
  const int h  = blockIdx.y;              // 0..2
  const int bz = blockIdx.z;              // 0..1023
  const int b  = bz & (B_ - 1);
  const int br = bz >> 9;                 // branch
  const int w  = b & (NW - 1);
  const size_t TQ = (size_t)B_ * NH * NP * HD;
  const half_t* Qp = Q  + (size_t)br * TQ;        // q from own modality
  const half_t* Kp = K  + (size_t)(1 - br) * TQ;  // k/v from the other one
  const half_t* Vp = Vt + (size_t)(1 - br) * TQ;

  __shared__ __align__(16) float  Ssm[64 * KT];   // 90112 B
  __shared__ __align__(16) half_t Pls[64 * KT];   // 45056 B
  __shared__ float invrow[64];

  const int tid  = threadIdx.x;
  const int lane = tid & 31;
  const int wv   = tid >> 5;
  const int hf   = lane >> 4;
  const int ln   = lane & 15;

  const size_t qkBase = ((size_t)b * NH + h) * NP;

  // Q A-tiles (reused across all 22 key tiles)
  v16h aq[4];
#pragma unroll
  for (int qt = 0; qt < 4; qt++) {
    int q = qb * 64 + qt * 16 + ln;               // < NP
    const half_t* p = Qp + (qkBase + q) * HD + 16 * hf;
    aq[qt] = pack2(*(const v8h*)p, *(const v8h*)(p + 8));
  }

  for (int kt = wv; kt < 22; kt += 8) {
    int key = kt * 16 + ln;
    v16h bk = *(const v16h*)(Kp + (qkBase + key) * HD + 16 * hf);
    int col = kt * 16 + ln;
#pragma unroll
    for (int qt = 0; qt < 4; qt++) {
      v8f acc = {};
      acc = wmma_f16(aq[qt], bk, acc);
#pragma unroll
      for (int r = 0; r < 8; r++) {
        int rowl = qt * 16 + r + 8 * hf;
        Ssm[rowl * KT + col] = acc[r];
      }
    }
  }
  __syncthreads();

  // softmax: 4 lanes per row; comb (bias+mask) added here with coalesced reads
  {
    int row = tid >> 2, sub = tid & 3;
    int q   = qb * 64 + row;
    int qc  = q < NTOK ? q : NTOK - 1;            // clamp for garbage pad rows
    const half_t* crow = comb + ((size_t)(w * NH + h) * NTOK + qc) * KT;
    float mx = -3.0e30f;
    for (int c = sub; c < KT; c += 4) {
      float v = Ssm[row * KT + c] + (float)crow[c];
      Ssm[row * KT + c] = v;
      mx = fmaxf(mx, v);
    }
    mx = fmaxf(mx, __shfl_xor(mx, 1));
    mx = fmaxf(mx, __shfl_xor(mx, 2));
    float s = 0.0f;
    for (int c = sub; c < KT; c += 4) {
      float e = __expf(Ssm[row * KT + c] - mx);
      s += e;
      Pls[row * KT + c] = (half_t)e;              // un-normalized probs
    }
    s += __shfl_xor(s, 1);
    s += __shfl_xor(s, 2);
    if (sub == 0) invrow[row] = 1.0f / s;
  }
  __syncthreads();

  // O = P * V : one 16x16 tile per wave, 11 WMMA K-steps
  {
    int qt = wv >> 1, dt = wv & 1;
    int d  = dt * 16 + ln;
    const half_t* vrow = Vp + (((size_t)b * NH + h) * HD + d) * NP;
    v8f acc = {};
#pragma unroll
    for (int kt = 0; kt < 11; kt++) {
      const int kbase = kt * 32;
      const half_t* pp = &Pls[(qt * 16 + ln) * KT + kbase + 16 * hf];
      v16h a  = pack2(*(const v8h*)pp, *(const v8h*)(pp + 8));
      v16h bv = *(const v16h*)(vrow + kbase + 16 * hf);
      acc = wmma_f16(a, bv, acc);
    }
    half_t* outp = attn_out + (size_t)br * B_ * NTOK * Cdim;
#pragma unroll
    for (int r = 0; r < 8; r++) {
      int rowl = qt * 16 + r + 8 * hf;
      int q    = qb * 64 + rowl;
      if (q < NTOK)
        outp[((size_t)b * NTOK + q) * Cdim + h * HD + d] = (half_t)(acc[r] * invrow[rowl]);
    }
  }
}

// ---------------------------------------------------------------------------
// Kernel 3: output projection. M = 2*512*343 = 351232 rows (16-divisible),
// K = 96, N = 96. f32 result straight into d_out (layout matches).
// 192 threads = 6 waves, one feature tile each.
// ---------------------------------------------------------------------------
__global__ __launch_bounds__(192) void k_proj(const half_t* __restrict__ ain,
                                              const half_t* __restrict__ ph,
                                              const float* __restrict__ bias,
                                              float* __restrict__ out) {
  const int mtile = blockIdx.x;
  __shared__ __align__(16) half_t As[16 * Cdim];
  const int tid = threadIdx.x;
  const uint32_t* src = (const uint32_t*)(ain + (size_t)mtile * 16 * Cdim);
  uint32_t* dst = (uint32_t*)As;
  for (int i = tid; i < (16 * Cdim) / 2; i += 192) dst[i] = src[i];
  __syncthreads();

  const int lane = tid & 31, wv = tid >> 5;
  const int hf = lane >> 4, ln = lane & 15;
  v16h a[3];
#pragma unroll
  for (int kc = 0; kc < 3; kc++) {
    const half_t* pr = &As[ln * Cdim + kc * 32 + 16 * hf];
    a[kc] = pack2(*(const v8h*)pr, *(const v8h*)(pr + 8));
  }
  const int f = wv * 16 + ln;
  v8f acc = {};
#pragma unroll
  for (int kc = 0; kc < 3; kc++) {
    v16h bm = *(const v16h*)(ph + (size_t)f * Cdim + kc * 32 + 16 * hf);
    acc = wmma_f16(a[kc], bm, acc);
  }
  const float bv = bias[f];
#pragma unroll
  for (int r = 0; r < 8; r++) {
    size_t row = (size_t)mtile * 16 + r + 8 * hf;
    out[row * Cdim + f] = acc[r] + bv;
  }
}

// ---------------------------------------------------------------------------
extern "C" void kernel_launch(void* const* d_in, const int* in_sizes, int n_in,
                              void* d_out, int out_size, void* d_ws, size_t ws_size,
                              hipStream_t stream) {
  (void)in_sizes; (void)n_in; (void)out_size; (void)ws_size;
  const float* x      = (const float*)d_in[0];
  const float* x1     = (const float*)d_in[1];
  const float* mask   = (const float*)d_in[2];
  const float* qkv_w  = (const float*)d_in[3];
  const float* qkv_b  = (const float*)d_in[4];
  const float* proj_w = (const float*)d_in[5];
  const float* proj_b = (const float*)d_in[6];
  const float* table  = (const float*)d_in[7];
  const int*   relidx = (const int*)d_in[8];

  const size_t TQ = (size_t)B_ * NH * NP * HD;   // halves per Q-like tensor
  half_t* p    = (half_t*)d_ws;
  half_t* Q    = p; p += 2 * TQ;
  half_t* K    = p; p += 2 * TQ;
  half_t* Vt   = p; p += 2 * TQ;
  half_t* comb = p; p += (size_t)NW * NH * NTOK * KT;
  half_t* aout = p; p += (size_t)2 * B_ * NTOK * Cdim;
  half_t* wh   = p; p += (size_t)WQN;
  half_t* ph   = p; p += (size_t)WPN;

  k_cvtw<<<dim3((WQN + WPN + 255) / 256), dim3(256), 0, stream>>>(qkv_w, proj_w, wh, ph);
  k_comb<<<dim3(NW * NH * NTOK), dim3(KT), 0, stream>>>(mask, table, relidx, comb);
  k_qkv <<<dim3(22, B_, 2),      dim3(288), 0, stream>>>(x, x1, wh, qkv_b, Q, K, Vt);
  k_attn<<<dim3(6, NH, 2 * B_),  dim3(256), 0, stream>>>(Q, K, Vt, comb, aout);
  k_proj<<<dim3((2 * B_ * NTOK) / 16), dim3(192), 0, stream>>>(aout, proj_w ? ph : ph, proj_b, (float*)d_out);
}
